// MFN_15522011807918
// MI455X (gfx1250) — matrix operand order repaired and checked by reading, not compiled, not run on hardware
//
#include <hip/hip_runtime.h>
#include <hip/hip_bf16.h>
#include <math.h>

typedef __attribute__((ext_vector_type(16))) _Float16 v16h;
typedef __attribute__((ext_vector_type(8)))  _Float16 v8h;
typedef __attribute__((ext_vector_type(8)))  float    v8f;

#define BN 2048
#define TN 50

__device__ __forceinline__ float sigf(float x) { return 1.0f / (1.0f + expf(-x)); }

// ---------------------------------------------------------------------------
// Fragment loads per CDNA5 ISA §7.12.2 (wave32).
// A 16x32 f16: lanes 0-15 -> M=lane, K = k0+0..7 and k0+16..23
//              lanes 16-31 -> M=lane-16, K = k0+8..15 and k0+24..31
// ---------------------------------------------------------------------------
__device__ __forceinline__ v16h load_a_frag(const _Float16* __restrict__ A, int lda,
                                            int row0, int k0, int lane) {
  const _Float16* p = A + (size_t)(row0 + (lane & 15)) * lda + k0 + ((lane >> 4) << 3);
  v8h lo = *(const v8h*)p;
  v8h hi = *(const v8h*)(p + 16);
  v16h r;
#pragma unroll
  for (int i = 0; i < 8; ++i) { r[i] = lo[i]; r[i + 8] = hi[i]; }
  return r;
}

// B 32x16 f16 (loaded as rows of W == columns of B):
// lanes 0-15 -> N=lane, K = k0+0..7 and k0+16..23 ; lanes 16-31 -> N=lane-16, K = k0+8..15 and k0+24..31
__device__ __forceinline__ v16h load_b_frag(const _Float16* __restrict__ W, int ldw,
                                            int col0, int k0, int lane) {
  const _Float16* p = W + (size_t)(col0 + (lane & 15)) * ldw + k0 + ((lane >> 4) << 3);
  v8h lo = *(const v8h*)p;
  v8h hi = *(const v8h*)(p + 16);
  v16h r;
#pragma unroll
  for (int i = 0; i < 8; ++i) { r[i] = lo[i]; r[i + 8] = hi[i]; }
  return r;
}

// ---------------------------------------------------------------------------
// Generic WMMA GEMM:  C[M x N] = act( [A1 | A2] @ W^T + bias )
// A segments row-major f16, W row-major f16 [N, K1+K2]. K1,K2 multiples of 32.
// WG = 256 threads = 8 waves; wave tile 32x32 (2x2 wmma); WG tile 128x64.
// grid = (N/64, M/128).
// ---------------------------------------------------------------------------
__global__ __launch_bounds__(256) void gemm_f16_wmma(
    const _Float16* __restrict__ A1, int lda1, int K1,
    const _Float16* __restrict__ A2, int lda2, int K2,
    const _Float16* __restrict__ W,  int ldw,
    const float* __restrict__ bias, int relu,
    float* __restrict__ Cf, _Float16* __restrict__ Ch, int ldc)
{
  const int lane = threadIdx.x & 31;
  const int wave = threadIdx.x >> 5;
  const int row0 = blockIdx.y * 128 + (wave >> 1) * 32;
  const int col0 = blockIdx.x * 64  + (wave & 1) * 32;

  v8f acc[2][2] = {};

  const _Float16* Asrc[2] = {A1, A2};
  const int ldas[2] = {lda1, lda2};
  const int Ks[2]   = {K1, K2};
  int kw = 0;
  for (int s = 0; s < 2; ++s) {
    const _Float16* A = Asrc[s];
    const int lda = ldas[s];
    const int K = Ks[s];
    for (int k0 = 0; k0 < K; k0 += 32) {
      v16h a0 = load_a_frag(A, lda, row0,      k0, lane);
      v16h a1 = load_a_frag(A, lda, row0 + 16, k0, lane);
      v16h b0 = load_b_frag(W, ldw, col0,      kw + k0, lane);
      v16h b1 = load_b_frag(W, ldw, col0 + 16, kw + k0, lane);
      acc[0][0] = __builtin_amdgcn_wmma_f32_16x16x32_f16(false, a0, false, b0, (short)0, acc[0][0], false, false);
      acc[0][1] = __builtin_amdgcn_wmma_f32_16x16x32_f16(false, a0, false, b1, (short)0, acc[0][1], false, false);
      acc[1][0] = __builtin_amdgcn_wmma_f32_16x16x32_f16(false, a1, false, b0, (short)0, acc[1][0], false, false);
      acc[1][1] = __builtin_amdgcn_wmma_f32_16x16x32_f16(false, a1, false, b1, (short)0, acc[1][1], false, false);
    }
    kw += K;
  }

  // C/D layout: lanes 0-15: N=lane, VGPR i -> M=i ; lanes 16-31: N=lane-16, M=i+8
  const int nn = lane & 15;
  const int mb = (lane >> 4) << 3;
#pragma unroll
  for (int mi = 0; mi < 2; ++mi) {
#pragma unroll
    for (int ni = 0; ni < 2; ++ni) {
      const int n = col0 + ni * 16 + nn;
      const float bv = bias ? bias[n] : 0.0f;
#pragma unroll
      for (int i = 0; i < 8; ++i) {
        const int m = row0 + mi * 16 + mb + i;
        float v = acc[mi][ni][i] + bv;
        if (relu) v = fmaxf(v, 0.0f);
        if (Cf) Cf[(size_t)m * ldc + n] = v;
        if (Ch) Ch[(size_t)m * ldc + n] = (_Float16)v;
      }
    }
  }
}

// ---------------------------------------------------------------------------
// Elementwise / setup kernels
// ---------------------------------------------------------------------------
__global__ void zero_u32(unsigned int* p, int n) {
  int i = blockIdx.x * blockDim.x + threadIdx.x;
  if (i < n) p[i] = 0u;
}

// Convert fp32 weight [rows, scols] -> f16 padded into dst[rows, dld] at col dcoff,
// writing wcols columns (zero pad beyond scols).
__global__ void conv_w(const float* __restrict__ src, _Float16* __restrict__ dst,
                       int rows, int scols, int dld, int dcoff, int wcols) {
  int idx = blockIdx.x * blockDim.x + threadIdx.x;
  int total = rows * wcols;
  if (idx >= total) return;
  int r = idx / wcols, c = idx % wcols;
  dst[(size_t)r * dld + dcoff + c] = (c < scols) ? (_Float16)src[(size_t)r * scols + c] : (_Float16)0.0f;
}

__global__ void vadd(const float* a, const float* b, float* o, int n) {
  int i = blockIdx.x * blockDim.x + threadIdx.x;
  if (i < n) o[i] = a[i] + b[i];
}

// Per-step input gather+convert: xcat[B,480] = [text(320 pad) | audio(96 pad) | video(64 pad)]
__global__ void xcat_k(const float* __restrict__ tx, const float* __restrict__ ax,
                       const float* __restrict__ vx, int t, _Float16* __restrict__ xc) {
  int idx = blockIdx.x * blockDim.x + threadIdx.x;
  if (idx >= BN * 480) return;
  int b = idx / 480, c = idx % 480;
  float v = 0.0f;
  if (c < 320)      { if (c < 300)            v = tx[((size_t)b * TN + t) * 300 + c]; }
  else if (c < 416) { int cc = c - 320; if (cc < 74) v = ax[((size_t)b * TN + t) * 74 + cc]; }
  else              { int cc = c - 416; if (cc < 35) v = vx[((size_t)b * TN + t) * 35 + cc]; }
  xc[idx] = (_Float16)v;
}

// LSTM cell nonlinearity. z[B,4H] (torch gate order i,f,g,o). c/h/h16 have ld=256
// with column offset `off`; cstar[B,512] gets prev c at off, new c at 256+off.
__global__ void lstm_elem(const float* __restrict__ z, float* __restrict__ c,
                          float* __restrict__ h, _Float16* __restrict__ h16,
                          _Float16* __restrict__ cstar, int H, int off) {
  int idx = blockIdx.x * blockDim.x + threadIdx.x;
  if (idx >= BN * H) return;
  int b = idx / H, j = idx % H;
  size_t zb = (size_t)b * 4 * H;
  float gi = z[zb + j], gf = z[zb + H + j], gg = z[zb + 2 * H + j], go = z[zb + 3 * H + j];
  float cold = c[(size_t)b * 256 + off + j];
  float cnew = sigf(gf) * cold + sigf(gi) * tanhf(gg);
  float hnew = sigf(go) * tanhf(cnew);
  c[(size_t)b * 256 + off + j]   = cnew;
  h[(size_t)b * 256 + off + j]   = hnew;
  h16[(size_t)b * 256 + off + j] = (_Float16)hnew;
  cstar[(size_t)b * 512 + off + j]       = (_Float16)cold;
  cstar[(size_t)b * 512 + 256 + off + j] = (_Float16)cnew;
}

// Fused row softmax (512) + attended = p * cstar, output f16
__global__ __launch_bounds__(256) void softmax_attend(const float* __restrict__ a1,
                                                      const _Float16* __restrict__ cstar,
                                                      _Float16* __restrict__ att) {
  __shared__ float red[256];
  const int b = blockIdx.x;
  const int t = threadIdx.x;
  const float x0 = a1[(size_t)b * 512 + t];
  const float x1 = a1[(size_t)b * 512 + 256 + t];
  red[t] = fmaxf(x0, x1);
  __syncthreads();
  for (int s = 128; s > 0; s >>= 1) {
    if (t < s) red[t] = fmaxf(red[t], red[t + s]);
    __syncthreads();
  }
  const float m = red[0];
  __syncthreads();
  const float e0 = expf(x0 - m), e1 = expf(x1 - m);
  red[t] = e0 + e1;
  __syncthreads();
  for (int s = 128; s > 0; s >>= 1) {
    if (t < s) red[t] += red[t + s];
    __syncthreads();
  }
  const float inv = 1.0f / red[0];
  att[(size_t)b * 512 + t]       = (_Float16)(e0 * inv * (float)cstar[(size_t)b * 512 + t]);
  att[(size_t)b * 512 + 256 + t] = (_Float16)(e1 * inv * (float)cstar[(size_t)b * 512 + 256 + t]);
}

// mem = sigmoid(g1o)*mem + sigmoid(g2o)*tanh(cpre)
__global__ void gate_mem(const float* __restrict__ g1o, const float* __restrict__ g2o,
                         const float* __restrict__ cpre, float* __restrict__ mem,
                         _Float16* __restrict__ mem16) {
  int idx = blockIdx.x * blockDim.x + threadIdx.x;
  if (idx >= BN * 128) return;
  float m2 = sigf(g1o[idx]) * mem[idx] + sigf(g2o[idx]) * tanhf(cpre[idx]);
  mem[idx] = m2;
  mem16[idx] = (_Float16)m2;
}

// last_hs = [h(256) | mem(128)] -> d_out + 2048
__global__ void assemble_lasths(const float* __restrict__ h, const float* __restrict__ mem,
                                float* __restrict__ out) {
  int idx = blockIdx.x * blockDim.x + threadIdx.x;
  if (idx >= BN * 384) return;
  int b = idx / 384, j = idx % 384;
  out[BN + idx] = (j < 256) ? h[(size_t)b * 256 + j] : mem[(size_t)b * 128 + (j - 256)];
}

// output[b] = to1[b,:] . w2 + b2
__global__ void out_final(const float* __restrict__ to1, const float* __restrict__ w2,
                          const float* __restrict__ b2, float* __restrict__ out) {
  int b = blockIdx.x * blockDim.x + threadIdx.x;
  if (b >= BN) return;
  float s = b2[0];
#pragma unroll 4
  for (int k = 0; k < 128; ++k) s += to1[(size_t)b * 128 + k] * w2[k];
  out[b] = s;
}

// ---------------------------------------------------------------------------
// Host launcher
// ---------------------------------------------------------------------------
static inline size_t aln(size_t x) { return (x + 255) & ~(size_t)255; }

static void gemm(hipStream_t st, const _Float16* A1, int lda1, int K1,
                 const _Float16* A2, int lda2, int K2,
                 const _Float16* W, int ldw, const float* bias, int relu,
                 float* Cf, _Float16* Ch, int N) {
  dim3 grid(N / 64, BN / 128);
  gemm_f16_wmma<<<grid, 256, 0, st>>>(A1, lda1, K1, A2, lda2, K2, W, ldw, bias, relu, Cf, Ch, N);
}

static inline dim3 g1d(int n) { return dim3((n + 255) / 256); }

extern "C" void kernel_launch(void* const* d_in, const int* in_sizes, int n_in,
                              void* d_out, int out_size, void* d_ws, size_t ws_size,
                              hipStream_t stream) {
  (void)in_sizes; (void)n_in; (void)out_size; (void)ws_size;

  // ---- inputs (setup_inputs dict order) ----
  const float* text_x  = (const float*)d_in[0];
  const float* audio_x = (const float*)d_in[1];
  const float* video_x = (const float*)d_in[2];
  const float* Wih_l = (const float*)d_in[3],  *Whh_l = (const float*)d_in[4];
  const float* bih_l = (const float*)d_in[5],  *bhh_l = (const float*)d_in[6];
  const float* Wih_a = (const float*)d_in[7],  *Whh_a = (const float*)d_in[8];
  const float* bih_a = (const float*)d_in[9],  *bhh_a = (const float*)d_in[10];
  const float* Wih_v = (const float*)d_in[11], *Whh_v = (const float*)d_in[12];
  const float* bih_v = (const float*)d_in[13], *bhh_v = (const float*)d_in[14];
  const float* at1w1 = (const float*)d_in[15], *at1b1 = (const float*)d_in[16];
  const float* at1w2 = (const float*)d_in[17], *at1b2 = (const float*)d_in[18];
  const float* at2w1 = (const float*)d_in[19], *at2b1 = (const float*)d_in[20];
  const float* at2w2 = (const float*)d_in[21], *at2b2 = (const float*)d_in[22];
  const float* g1w1  = (const float*)d_in[23], *g1b1  = (const float*)d_in[24];
  const float* g1w2  = (const float*)d_in[25], *g1b2  = (const float*)d_in[26];
  const float* g2w1  = (const float*)d_in[27], *g2b1  = (const float*)d_in[28];
  const float* g2w2  = (const float*)d_in[29], *g2b2  = (const float*)d_in[30];
  const float* ow1   = (const float*)d_in[31], *ob1   = (const float*)d_in[32];
  const float* ow2   = (const float*)d_in[33], *ob2   = (const float*)d_in[34];

  float* out = (float*)d_out;

  // ---- workspace layout ----
  char* ws = (char*)d_ws;
  size_t off = 0;
  auto alloc = [&](size_t bytes) -> char* { char* p = ws + off; off = aln(off + bytes); return p; };

  // f16 weights (K padded to multiples of 32)
  _Float16* Wl16   = (_Float16*)alloc((size_t)512 * 448 * 2); // [x(320)|h(128)]
  _Float16* Wa16   = (_Float16*)alloc((size_t)256 * 160 * 2); // [x(96)|h(64)]
  _Float16* Wv16   = (_Float16*)alloc((size_t)256 * 128 * 2); // [x(64)|h(64)]
  _Float16* Wat1a  = (_Float16*)alloc((size_t)256 * 512 * 2);
  _Float16* Wat1b  = (_Float16*)alloc((size_t)512 * 256 * 2);
  _Float16* Wat2a  = (_Float16*)alloc((size_t)256 * 512 * 2);
  _Float16* Wat2b  = (_Float16*)alloc((size_t)128 * 256 * 2);
  _Float16* Wg1a   = (_Float16*)alloc((size_t)128 * 640 * 2);
  _Float16* Wg1b   = (_Float16*)alloc((size_t)128 * 128 * 2);
  _Float16* Wg2a   = (_Float16*)alloc((size_t)128 * 640 * 2);
  _Float16* Wg2b   = (_Float16*)alloc((size_t)128 * 128 * 2);
  _Float16* Wout1  = (_Float16*)alloc((size_t)128 * 384 * 2);
  float* bl = (float*)alloc(512 * 4);
  float* ba = (float*)alloc(256 * 4);
  float* bv = (float*)alloc(256 * 4);

  // persistent state (contiguous; zeroed below)
  char* state = ws + off;
  float*    h      = (float*)   alloc((size_t)BN * 256 * 4);
  float*    c      = (float*)   alloc((size_t)BN * 256 * 4);
  float*    mem    = (float*)   alloc((size_t)BN * 128 * 4);
  _Float16* h16    = (_Float16*)alloc((size_t)BN * 256 * 2);
  _Float16* mem16  = (_Float16*)alloc((size_t)BN * 128 * 2);
  _Float16* cstar  = (_Float16*)alloc((size_t)BN * 512 * 2);
  size_t state_bytes = (size_t)(ws + off - state);

  // per-step scratch
  _Float16* xcat = (_Float16*)alloc((size_t)BN * 480 * 2);
  float*    zl   = (float*)   alloc((size_t)BN * 512 * 4);
  float*    za   = (float*)   alloc((size_t)BN * 256 * 4);
  float*    zv   = (float*)   alloc((size_t)BN * 256 * 4);
  _Float16* t1   = (_Float16*)alloc((size_t)BN * 256 * 2);
  float*    a1   = (float*)   alloc((size_t)BN * 512 * 4);
  _Float16* att  = (_Float16*)alloc((size_t)BN * 512 * 2);
  _Float16* t2   = (_Float16*)alloc((size_t)BN * 256 * 2);
  float*    cpre = (float*)   alloc((size_t)BN * 128 * 4);
  _Float16* tg1  = (_Float16*)alloc((size_t)BN * 128 * 2);
  float*    g1o  = (float*)   alloc((size_t)BN * 128 * 4);
  _Float16* tg2  = (_Float16*)alloc((size_t)BN * 128 * 2);
  float*    g2o  = (float*)   alloc((size_t)BN * 128 * 4);
  float*    to1  = (float*)   alloc((size_t)BN * 128 * 4);

  // ---- weight conversion (once per launch) ----
  auto cw = [&](const float* s, _Float16* d, int rows, int scols, int dld, int dcoff, int wcols) {
    conv_w<<<g1d(rows * wcols), 256, 0, stream>>>(s, d, rows, scols, dld, dcoff, wcols);
  };
  cw(Wih_l, Wl16, 512, 300, 448, 0, 320);  cw(Whh_l, Wl16, 512, 128, 448, 320, 128);
  cw(Wih_a, Wa16, 256,  74, 160, 0,  96);  cw(Whh_a, Wa16, 256,  64, 160,  96,  64);
  cw(Wih_v, Wv16, 256,  35, 128, 0,  64);  cw(Whh_v, Wv16, 256,  64, 128,  64,  64);
  cw(at1w1, Wat1a, 256, 512, 512, 0, 512); cw(at1w2, Wat1b, 512, 256, 256, 0, 256);
  cw(at2w1, Wat2a, 256, 512, 512, 0, 512); cw(at2w2, Wat2b, 128, 256, 256, 0, 256);
  cw(g1w1,  Wg1a,  128, 640, 640, 0, 640); cw(g1w2,  Wg1b,  128, 128, 128, 0, 128);
  cw(g2w1,  Wg2a,  128, 640, 640, 0, 640); cw(g2w2,  Wg2b,  128, 128, 128, 0, 128);
  cw(ow1,   Wout1, 128, 384, 384, 0, 384);
  vadd<<<g1d(512), 256, 0, stream>>>(bih_l, bhh_l, bl, 512);
  vadd<<<g1d(256), 256, 0, stream>>>(bih_a, bhh_a, ba, 256);
  vadd<<<g1d(256), 256, 0, stream>>>(bih_v, bhh_v, bv, 256);

  // ---- zero persistent state ----
  int state_words = (int)(state_bytes / 4);
  zero_u32<<<g1d(state_words), 256, 0, stream>>>((unsigned int*)state, state_words);

  // ---- time loop ----
  for (int t = 0; t < TN; ++t) {
    xcat_k<<<g1d(BN * 480), 256, 0, stream>>>(text_x, audio_x, video_x, t, xcat);

    // LSTM pre-activations z = [x|h] @ W^T + (bih+bhh)
    gemm(stream, xcat,        480, 320, h16,        256, 128, Wl16, 448, bl, 0, zl, nullptr, 512);
    gemm(stream, xcat + 320,  480,  96, h16 + 128,  256,  64, Wa16, 160, ba, 0, za, nullptr, 256);
    gemm(stream, xcat + 416,  480,  64, h16 + 192,  256,  64, Wv16, 128, bv, 0, zv, nullptr, 256);

    lstm_elem<<<g1d(BN * 128), 256, 0, stream>>>(zl, c, h, h16, cstar, 128,   0);
    lstm_elem<<<g1d(BN *  64), 256, 0, stream>>>(za, c, h, h16, cstar,  64, 128);
    lstm_elem<<<g1d(BN *  64), 256, 0, stream>>>(zv, c, h, h16, cstar,  64, 192);

    // attention MLP 1: 512 -> 256 (relu) -> 512, then softmax * cstar
    gemm(stream, cstar, 512, 512, nullptr, 0, 0, Wat1a, 512, at1b1, 1, nullptr, t1, 256);
    gemm(stream, t1,    256, 256, nullptr, 0, 0, Wat1b, 256, at1b2, 0, a1, nullptr, 512);
    softmax_attend<<<BN, 256, 0, stream>>>(a1, cstar, att);

    // attention MLP 2: 512 -> 256 (relu) -> 128 (tanh applied in gate_mem)
    gemm(stream, att, 512, 512, nullptr, 0, 0, Wat2a, 512, at2b1, 1, nullptr, t2, 256);
    gemm(stream, t2,  256, 256, nullptr, 0, 0, Wat2b, 256, at2b2, 0, cpre, nullptr, 128);

    // gates: [attended|mem] (640) -> 128 (relu) -> 128
    gemm(stream, att, 512, 512, mem16, 128, 128, Wg1a, 640, g1b1, 1, nullptr, tg1, 128);
    gemm(stream, tg1, 128, 128, nullptr, 0, 0,   Wg1b, 128, g1b2, 0, g1o, nullptr, 128);
    gemm(stream, att, 512, 512, mem16, 128, 128, Wg2a, 640, g2b1, 1, nullptr, tg2, 128);
    gemm(stream, tg2, 128, 128, nullptr, 0, 0,   Wg2b, 128, g2b2, 0, g2o, nullptr, 128);

    gate_mem<<<g1d(BN * 128), 256, 0, stream>>>(g1o, g2o, cpre, mem, mem16);
  }

  // ---- output head: [h(256)|mem(128)] -> 128 (relu) -> 1 ----
  gemm(stream, h16, 256, 256, mem16, 128, 128, Wout1, 384, ob1, 1, to1, nullptr, 128);
  out_final<<<g1d(BN), 256, 0, stream>>>(to1, ow2, ob2, out);
  assemble_lasths<<<g1d(BN * 384), 256, 0, stream>>>(h, mem, out);
}
